// AutoregressiveLSTMCell_57561151701535
// MI455X (gfx1250) — matrix operand order repaired and checked by reading, not compiled, not run on hardware
//
#include <hip/hip_runtime.h>

// Problem dims (fixed by the reference)
#define Bz 256
#define Tz 256
#define Dz 256
#define Hz 1024
#define Oz 256
#define G4H (4 * Hz)   // 4096 gate columns
#define KX  (Dz + Oz)  // 512  rows of Wx

#define CHUNK_K 128            // K elements staged per TDM chunk
#define NCHUNK  12             // 4 (x|y via Wx) + 8 (h via Wh)
#define LDS_PITCH (CHUNK_K + 8) // 136 elems = 272B row pitch (TDM pad: 4dw / 64dw)

typedef __bf16 bf16;
typedef __attribute__((ext_vector_type(16))) __bf16 v16bf;
typedef __attribute__((ext_vector_type(8)))  float  v8f;
typedef __attribute__((ext_vector_type(4)))  unsigned int v4u;
typedef __attribute__((ext_vector_type(8)))  int v8i;
typedef __attribute__((ext_vector_type(4)))  int v4i;

union FragU { uint4 u[2]; v16bf v; };

// 16-bit A-matrix 16x32 fragment as loaded here, contiguous k order (wave32):
//   lane L(0..15)=row M, half=0 -> K 0..15
//   lane 16+L     =row M, half=1 -> K 16..31
__device__ __forceinline__ v16bf load_a_frag(const bf16* __restrict__ row, int half, int k0) {
    FragU f;
    f.u[0] = *(const uint4*)(row + k0 + half * 16);
    f.u[1] = *(const uint4*)(row + k0 + half * 16 + 8);
    return f.v;
}

// 16-bit B-matrix 32x16 fragment, same contiguous k order as A (wave32): lanes 0..15 = col N, K 0..15 ; lanes 16..31 = K 16..31
// From an LDS row (pitch LDS_PITCH) holding one weight column's K-chunk.
__device__ __forceinline__ v16bf load_b_lds(const bf16* row, int half, int kk) {
    const uint4* p = (const uint4*)(row + kk + half * 16);
    FragU f;
    f.u[0] = p[0];
    f.u[1] = p[1];
    return f.v;
}

__device__ __forceinline__ float sigmoid_fast(float x) {
    return 1.0f / (1.0f + __expf(-x));
}
__device__ __forceinline__ float tanh_fast(float x) {
    float e = __expf(2.0f * x);
    return (e - 1.0f) / (e + 1.0f);
}

#define WMMA_BF16(A, B, C) \
    __builtin_amdgcn_wmma_f32_16x16x32_bf16(false, (A), false, (B), (short)0, (C), false, false)

// ---------------------------------------------------------------------------
// TDM: DMA a 2-D tile (16 rows x CHUNK_K bf16, row stride = rowlen elems) from
// global into LDS at lds_byte_addr, inserting 4 dwords of LDS padding after
// every 64 dwords (one 128-elem row) -> LDS row pitch 136 elems.
// Descriptor packing per CDNA5 ISA §8.3/8.4 (D# group0/group1).
// ---------------------------------------------------------------------------
__device__ __forceinline__ void tdm_load_2d(unsigned lds_byte_addr,
                                            const bf16* gptr,
                                            unsigned rowlen_elems) {
    const unsigned long long ga = (unsigned long long)(uintptr_t)gptr;
    v4u g0;
    g0[0] = 1u;                                   // count=1, no gather
    g0[1] = lds_byte_addr;                        // lds_addr
    g0[2] = (unsigned)(ga & 0xffffffffu);         // global_addr[31:0]
    g0[3] = (unsigned)((ga >> 32) & 0x1ffffffu)   // global_addr[56:32]
          | (2u << 30);                           // type=2 ("image")

    const unsigned td0 = rowlen_elems;            // tensor_dim0
    const unsigned td1 = 65536u;                  // tensor_dim1 (generous OOB bound)
    v8i g1;
    g1[0] = (int)((1u << 16)                      // data_size = 2 bytes
                | (1u << 20)                      // pad_enable
                | (5u << 22)                      // pad_interval: 64 dwords
                | (3u << 25));                    // pad_amount:   4 dwords
    g1[1] = (int)((td0 & 0xffffu) << 16);                         // dim0 lo16 @bits[63:48]
    g1[2] = (int)(((td0 >> 16) & 0xffffu) | ((td1 & 0xffffu) << 16));
    g1[3] = (int)(((td1 >> 16) & 0xffffu) | ((unsigned)CHUNK_K << 16)); // tile_dim0
    g1[4] = 16;                                   // tile_dim1 = 16 rows, tile_dim2 = 0
    g1[5] = (int)rowlen_elems;                    // tensor_dim0_stride lo32 (elems)
    g1[6] = 0;                                    // stride hi16 | dim1_stride lo16
    g1[7] = 0;                                    // dim1_stride hi32

    v4i gz = {};
#if defined(__clang_major__) && (__clang_major__ >= 23)
    v8i gz8 = {};
    __builtin_amdgcn_tensor_load_to_lds(g0, g1, gz, gz, gz8, 0);
#else
    __builtin_amdgcn_tensor_load_to_lds(g0, g1, gz, gz, 0);
#endif
}

// ---------------------------------------------------------------------------
// Kernel A: one LSTM step.  z = [x_t|y]@Wx + h@Wh ; gates ; c,h update.
// Block: 8 waves = 128 batch rows x 16 hidden cols (x4 gates).
//   B (weights) double-buffered in LDS via TDM, shared by all 8 waves.
//   A ([x|y|h] rows) streamed per-wave straight from global (L2-resident).
// Grid: 2 M-groups x 64 hidden tiles = 128 blocks x 256 threads.
// ---------------------------------------------------------------------------
__global__ __launch_bounds__(256) void lstm_step_kernel(
    const bf16* __restrict__ xbf,    // [B,T,D] bf16
    const bf16* __restrict__ ybf,    // [B,O]   bf16 (prev dense output)
    const bf16* __restrict__ hprev,  // [B,H]   bf16
    bf16*       __restrict__ hnext,  // [B,H]   bf16
    float*      __restrict__ cstate, // [B,H]   f32
    const bf16* __restrict__ WxT,    // [4H,KX] bf16 (transposed)
    const bf16* __restrict__ WhT,    // [4H,H]  bf16 (transposed)
    const float* __restrict__ bias,  // [4H]
    int t)
{
    __shared__ bf16 slab[2][4][16][LDS_PITCH];    // 34.8 KB, double buffered

    const int wave = threadIdx.x >> 5;
    const int lane = threadIdx.x & 31;
    const int half = lane >> 4;
    const int l15  = lane & 15;

    const int mgrp = blockIdx.x & 1;              // which half of the batch
    const int n0   = (blockIdx.x >> 1) * 16;      // hidden tile
    const int m0   = mgrp * 128 + wave * 16;      // this wave's batch tile
    const int ncol = n0 + l15;

    const int arow = m0 + l15;
    const bf16* ax = xbf   + ((size_t)arow * Tz + t) * Dz;
    const bf16* ay = ybf   + (size_t)arow * Oz;
    const bf16* ah = hprev + (size_t)arow * Hz;

    const unsigned lds_base = (unsigned)(uintptr_t)(&slab[0][0][0][0]);
    const unsigned buf_bytes = 4u * 16u * LDS_PITCH * 2u;
    const unsigned gate_bytes = 16u * LDS_PITCH * 2u;

    // ---- TDM issue helper (uniform values; wave 0 only issues) ----
    auto issue_chunk = [&](int c, int buf) {
        const bf16* wbase;
        unsigned rowlen, kw;
        if (c < 4) { wbase = WxT; rowlen = KX; kw = (unsigned)c * CHUNK_K; }
        else       { wbase = WhT; rowlen = Hz; kw = (unsigned)(c - 4) * CHUNK_K; }
        #pragma unroll
        for (int g = 0; g < 4; ++g) {
            const bf16* gp = wbase + (size_t)(g * Hz + n0) * rowlen + kw;
            tdm_load_2d(lds_base + (unsigned)buf * buf_bytes + (unsigned)g * gate_bytes,
                        gp, rowlen);
        }
    };

    v8f acc0 = {}, acc1 = {}, acc2 = {}, acc3 = {};

    if (wave == 0) issue_chunk(0, 0);             // prologue

    for (int c = 0; c < NCHUNK; ++c) {
        if (wave == 0) {
            if (c + 1 < NCHUNK) {
                issue_chunk(c + 1, (c + 1) & 1);  // prefetch next chunk
                __builtin_amdgcn_s_wait_tensorcnt(4);  // chunk c landed (in-order)
            } else {
                __builtin_amdgcn_s_wait_tensorcnt(0);
            }
        }
        __syncthreads();                          // LDS chunk c visible to all waves

        const int cb = c & 1;
        const bf16* abase = (c < 2) ? (ax + c * CHUNK_K)
                          : (c < 4) ? (ay + (c - 2) * CHUNK_K)
                                    : (ah + (c - 4) * CHUNK_K);
        const bf16* b0 = &slab[cb][0][l15][0];
        const bf16* b1 = &slab[cb][1][l15][0];
        const bf16* b2 = &slab[cb][2][l15][0];
        const bf16* b3 = &slab[cb][3][l15][0];
        __builtin_prefetch((const void*)(abase + CHUNK_K), 0, 0);

        #pragma unroll
        for (int kk = 0; kk < CHUNK_K; kk += 32) {
            v16bf a = load_a_frag(abase, half, kk);
            acc0 = WMMA_BF16(a, load_b_lds(b0, half, kk), acc0);
            acc1 = WMMA_BF16(a, load_b_lds(b1, half, kk), acc1);
            acc2 = WMMA_BF16(a, load_b_lds(b2, half, kk), acc2);
            acc3 = WMMA_BF16(a, load_b_lds(b3, half, kk), acc3);
        }
        __syncthreads();                          // done reading before next overwrite
    }

    // Epilogue: C/D layout -> lane owns N = ncol, M = m0 + half*8 + r
    const float bi  = bias[ncol];
    const float bf_ = bias[Hz + ncol];
    const float bg  = bias[2 * Hz + ncol];
    const float bo  = bias[3 * Hz + ncol];
    #pragma unroll
    for (int r = 0; r < 8; ++r) {
        const int m = m0 + half * 8 + r;
        const size_t idx = (size_t)m * Hz + ncol;
        const float iv = sigmoid_fast(acc0[r] + bi);
        const float fv = sigmoid_fast(acc1[r] + bf_);
        const float gv = tanh_fast(acc2[r] + bg);
        const float ov = sigmoid_fast(acc3[r] + bo);
        const float cn = fv * cstate[idx] + iv * gv;
        cstate[idx] = cn;
        hnext[idx]  = (bf16)(ov * tanh_fast(cn));
    }
}

// ---------------------------------------------------------------------------
// Kernel B: y = tanh(h @ Wd + bd).  M=256, N=256, K=1024.  Direct-from-global
// WMMA (weights are tiny and L2-resident).  32 blocks x 8 waves.
// ---------------------------------------------------------------------------
__global__ __launch_bounds__(256) void dense_step_kernel(
    const bf16* __restrict__ h,     // [B,H] bf16 (just-computed)
    const bf16* __restrict__ WdT,   // [O,H] bf16 (transposed)
    const float* __restrict__ bd,   // [O]
    bf16*  __restrict__ ybf,        // [B,O] bf16 feedback
    float* __restrict__ out,        // [B,T,O] f32
    int t)
{
    const int wave = threadIdx.x >> 5;
    const int lane = threadIdx.x & 31;
    const int half = lane >> 4;
    const int l15  = lane & 15;

    const int w  = blockIdx.x * 8 + wave;  // 0..255
    const int m0 = (w & 15) * 16;
    const int n0 = (w >> 4) * 16;

    const bf16* ah   = h   + (size_t)(m0 + l15) * Hz;
    const bf16* bcol = WdT + (size_t)(n0 + l15) * Hz;

    v8f acc = {};
    #pragma unroll 4
    for (int k = 0; k < Hz; k += 32) {
        FragU f;
        const uint4* p = (const uint4*)(bcol + k + half * 16);
        f.u[0] = p[0];
        f.u[1] = p[1];
        acc = WMMA_BF16(load_a_frag(ah, half, k), f.v, acc);
    }

    const int n = n0 + l15;
    const float bb = bd[n];
    #pragma unroll
    for (int r = 0; r < 8; ++r) {
        const int m = m0 + half * 8 + r;
        const float v = tanh_fast(acc[r] + bb);
        ybf[(size_t)m * Oz + n] = (bf16)v;
        out[((size_t)m * Tz + t) * Oz + n] = v;
    }
}

// ---------------------------------------------------------------------------
// Setup kernels (once per launch; cost amortized over ~860 GFLOP of WMMA)
// ---------------------------------------------------------------------------
__global__ void f32_to_bf16_kernel(const float* __restrict__ src, bf16* __restrict__ dst, size_t n) {
    size_t i = (size_t)blockIdx.x * blockDim.x + threadIdx.x;
    const size_t stride = (size_t)gridDim.x * blockDim.x;
    for (; i < n; i += stride) dst[i] = (bf16)src[i];
}

// src[K][N] f32 -> dst[N][K] bf16  (coalesced on the write side)
__global__ void transpose_bf16_kernel(const float* __restrict__ src, bf16* __restrict__ dst,
                                      int K, int N) {
    const size_t total = (size_t)K * N;
    size_t i = (size_t)blockIdx.x * blockDim.x + threadIdx.x;
    const size_t stride = (size_t)gridDim.x * blockDim.x;
    for (; i < total; i += stride) {
        const int n = (int)(i / K);
        const int k = (int)(i % K);
        dst[i] = (bf16)src[(size_t)k * N + n];
    }
}

__global__ void zero_u32_kernel(unsigned int* __restrict__ p, size_t n) {
    size_t i = (size_t)blockIdx.x * blockDim.x + threadIdx.x;
    const size_t stride = (size_t)gridDim.x * blockDim.x;
    for (; i < n; i += stride) p[i] = 0u;
}

// ---------------------------------------------------------------------------
extern "C" void kernel_launch(void* const* d_in, const int* in_sizes, int n_in,
                              void* d_out, int out_size, void* d_ws, size_t ws_size,
                              hipStream_t stream) {
    (void)in_sizes; (void)n_in; (void)out_size; (void)ws_size;

    const float* x  = (const float*)d_in[0];  // [B,T,D]
    const float* Wx = (const float*)d_in[1];  // [D+O, 4H]
    const float* Wh = (const float*)d_in[2];  // [H, 4H]
    const float* b  = (const float*)d_in[3];  // [4H]
    const float* Wd = (const float*)d_in[4];  // [H, O]
    const float* bd = (const float*)d_in[5];  // [O]
    float* out = (float*)d_out;               // [B,T,O]

    // Workspace carve-up (~46.5 MiB total)
    char* ws = (char*)d_ws;
    bf16* WxT = (bf16*)ws; ws += (size_t)G4H * KX * sizeof(bf16);      // 4 MiB
    bf16* WhT = (bf16*)ws; ws += (size_t)G4H * Hz * sizeof(bf16);      // 8 MiB
    bf16* WdT = (bf16*)ws; ws += (size_t)Oz * Hz * sizeof(bf16);       // 0.5 MiB
    bf16* xbf = (bf16*)ws; ws += (size_t)Bz * Tz * Dz * sizeof(bf16);  // 32 MiB
    // state block (zeroed together): h0, h1, y (bf16) then c (f32)
    char* state = ws;
    bf16* h0  = (bf16*)ws; ws += (size_t)Bz * Hz * sizeof(bf16);       // 0.5 MiB
    bf16* h1  = (bf16*)ws; ws += (size_t)Bz * Hz * sizeof(bf16);       // 0.5 MiB
    bf16* ybf = (bf16*)ws; ws += (size_t)Bz * Oz * sizeof(bf16);       // 128 KiB
    float* cst = (float*)ws; ws += (size_t)Bz * Hz * sizeof(float);    // 1 MiB
    const size_t state_u32 = (size_t)(ws - state) / 4;

    // Precision / layout prep
    f32_to_bf16_kernel<<<2048, 256, 0, stream>>>(x, xbf, (size_t)Bz * Tz * Dz);
    transpose_bf16_kernel<<<2048, 256, 0, stream>>>(Wx, WxT, KX, G4H);
    transpose_bf16_kernel<<<2048, 256, 0, stream>>>(Wh, WhT, Hz, G4H);
    transpose_bf16_kernel<<<512, 256, 0, stream>>>(Wd, WdT, Hz, Oz);
    zero_u32_kernel<<<1024, 256, 0, stream>>>((unsigned int*)state, state_u32);

    // Sequential recurrence: stream ordering provides the dependency chain.
    for (int t = 0; t < Tz; ++t) {
        const bf16* hp = (t & 1) ? h1 : h0;
        bf16*       hn = (t & 1) ? h0 : h1;
        lstm_step_kernel<<<128, 256, 0, stream>>>(xbf, ybf, hp, hn, cst,
                                                  WxT, WhT, b, t);
        dense_step_kernel<<<32, 256, 0, stream>>>(hn, WdT, bd, ybf, out, t);
    }
}
